// LocalCrossAttention_30219389894967
// MI455X (gfx1250) — matrix operand order repaired and checked by reading, not compiled, not run on hardware
//
#include <hip/hip_runtime.h>
#include <hip/hip_bf16.h>
#include <stdint.h>

// ---------------------------------------------------------------------------
// LocalCrossAttention for MI455X (gfx1250), wave32 + WMMA bf16.
//
// Shapes (fixed): B=2 N=4096 K=32 H=8 D=64, DIM_Q=DIM_OUT=512, DIM_C=256,
// INNER=512, SCALE=0.125. Tokens T = B*N = 8192.
//
// Algebra: attention is linear in K/V -> fold K/V projections through k=32:
//   u[t,h,c]  = sum_d q[t,h,d] * Wk[c, h*64+d]          (replaces K-proj)
//   s[t,h,k]  = SCALE * sum_c C[t,k,c]*u[t,h,c] + bias;  attn = softmax_k
//   w[t,h,c]  = sum_k attn[t,h,k] * C[t,k,c]
//   o[t,h,d]  = sum_c w[t,h,c] * Wv[c, h*64+d]          (replaces V-proj)
//   y         = o_flat @ Wo + bo
// ~10x FLOP reduction; context (268 MB) read exactly once -> memory bound
// (~13 us floor at 23.3 TB/s). GEMMs in bf16 WMMA with f32 accumulate.
//
// This revision: ping-pong double buffering via unroll-by-2 (buffers are
// reloaded in place, never copied) -> removes the v_dual_mov walls and the
// WMMA->VALU hazard v_nops seen in round 2's inner loop.
// ---------------------------------------------------------------------------

typedef __attribute__((ext_vector_type(16))) __bf16 v16bf;
typedef __attribute__((ext_vector_type(8)))  float  v8f;

#define DEVINL __device__ __forceinline__

// A fragment, 16x32 bf16, row-major source, leading dim lda.
// Contiguous k order used here: lane L: row m=L%16; lanes 0-15 hold K=0..15,
// lanes 16-31 hold K=16..31 (matches load_bT_bf16).
DEVINL v16bf load_a_bf16(const __bf16* A, int lda) {
  const int lane = threadIdx.x & 31;
  const int m  = lane & 15;
  const int kb = (lane < 16) ? 0 : 16;
  const __bf16* p = A + m * lda + kb;
  v16bf a;
#pragma unroll
  for (int i = 0; i < 8; ++i) a[i] = p[i];
#pragma unroll
  for (int i = 0; i < 8; ++i) a[i + 8] = p[i + 8];
  return a;
}

// B fragment, 32x16 bf16, from TRANSPOSED (N-major [N x K], ld=ldb) source so
// each lane reads 16 contiguous K values: lane L: col n=L%16,
// K = (L<16 ? 0 : 16) + 0..15.
DEVINL v16bf load_bT_bf16(const __bf16* BT, int ldb) {
  const int lane = threadIdx.x & 31;
  const int n  = lane & 15;
  const int kb = (lane < 16) ? 0 : 16;
  const __bf16* p = BT + n * ldb + kb;
  v16bf b;
#pragma unroll
  for (int i = 0; i < 16; ++i) b[i] = p[i];
  return b;
}

DEVINL v8f wmma_bf16f32(v16bf a, v16bf b, v8f c) {
  // v_wmma_f32_16x16x32_bf16 : (neg_a, A, neg_b, B, c_mod, C, reuse_a, reuse_b)
  return __builtin_amdgcn_wmma_f32_16x16x32_bf16(false, a, false, b,
                                                 (short)0, c, false, false);
}

// C/D 16x16 f32 layout: VGPR r -> row r (lanes 0-15) / row r+8 (lanes 16-31).
DEVINL void store_d_bf16(__bf16* D, int ldd, v8f c) {
  const int lane = threadIdx.x & 31;
  const int n  = lane & 15;
  const int mb = (lane < 16) ? 0 : 8;
#pragma unroll
  for (int r = 0; r < 8; ++r) D[(mb + r) * ldd + n] = (__bf16)c[r];
}

// -------------------------- prep kernels -----------------------------------
__global__ __launch_bounds__(256) void k_cvt_bf16(const float* __restrict__ s,
                                                  __bf16* __restrict__ d, int n) {
  int i = blockIdx.x * 256 + threadIdx.x;
  if (i < n) d[i] = (__bf16)s[i];
}

// src [rows x cols] f32 row-major -> dst [cols x rows] bf16
__global__ __launch_bounds__(256) void k_transpose_bf16(const float* __restrict__ s,
                                                        __bf16* __restrict__ d,
                                                        int rows, int cols) {
  int i = blockIdx.x * 256 + threadIdx.x;
  if (i < rows * cols) {
    int r = i / cols, c = i - r * cols;
    d[c * rows + r] = (__bf16)s[i];
  }
}

// --------------- K1: Q = X @ Wq   (8192x512 @ 512x512) ---------------------
// 16 tokens/block, 8 waves; wave w owns N cols [w*64, w*64+64).
// A tile staged once in LDS (16 KB, 8-wave reuse); B ping-pong via unroll-2.
__global__ __launch_bounds__(256) void k_gemm_q(const __bf16* __restrict__ Xb,
                                                const __bf16* __restrict__ WqT,
                                                __bf16* __restrict__ Qb) {
  const int w  = threadIdx.x >> 5;
  const int t0 = blockIdx.x * 16;

  __shared__ uint4 Ash4[1024];                     // 16 KB: A tile 16x512 bf16
  __bf16* Ash = (__bf16*)Ash4;
  {
    const uint4* src = (const uint4*)(Xb + (size_t)t0 * 512);
    for (int i = threadIdx.x; i < 1024; i += 256) Ash4[i] = src[i];
  }
  __syncthreads();

  const __bf16* Bbase = WqT + (size_t)(w * 64) * 512;
  v8f z = {};
  v8f acc[4] = {z, z, z, z};

  v16bf b0[4], b1[4];
#pragma unroll
  for (int f = 0; f < 4; ++f) b0[f] = load_bT_bf16(Bbase + (size_t)(f * 16) * 512, 512);

#pragma unroll
  for (int kk = 0; kk < 512; kk += 64) {
#pragma unroll
    for (int f = 0; f < 4; ++f)
      b1[f] = load_bT_bf16(Bbase + (size_t)(f * 16) * 512 + kk + 32, 512);
    v16bf a0 = load_a_bf16(Ash + kk, 512);
#pragma unroll
    for (int f = 0; f < 4; ++f) acc[f] = wmma_bf16f32(a0, b0[f], acc[f]);
    if (kk + 64 < 512) {
#pragma unroll
      for (int f = 0; f < 4; ++f)
        b0[f] = load_bT_bf16(Bbase + (size_t)(f * 16) * 512 + kk + 64, 512);
    }
    v16bf a1 = load_a_bf16(Ash + kk + 32, 512);
#pragma unroll
    for (int f = 0; f < 4; ++f) acc[f] = wmma_bf16f32(a1, b1[f], acc[f]);
  }
#pragma unroll
  for (int f = 0; f < 4; ++f)
    store_d_bf16(Qb + (size_t)t0 * 512 + w * 64 + f * 16, 512, acc[f]);
}

// --------------- K2: U_h = Q_h @ Wk_h^T  per head --------------------------
// U[t,h,c] = sum_d Q[t,h*64+d]*Wk[c,h*64+d]; B^T operand is Wk itself (rows=c,
// contiguous d). K=64 -> 2 chunks; all 6 fragment loads hoisted before WMMAs.
__global__ __launch_bounds__(256) void k_gemm_u(const __bf16* __restrict__ Qb,
                                                const __bf16* __restrict__ Wkb,
                                                __bf16* __restrict__ Ub) {
  const int w  = threadIdx.x >> 5;
  const int h  = blockIdx.y;
  const int t0 = blockIdx.x * 16;
  const int n0 = w * 32;                           // 2 n-frags per wave

  const __bf16* Abase = Qb + (size_t)t0 * 512 + h * 64;
  v16bf a0 = load_a_bf16(Abase, 512);
  v16bf a1 = load_a_bf16(Abase + 32, 512);
  v16bf b00 = load_bT_bf16(Wkb + (size_t)n0 * 512 + h * 64, 512);
  v16bf b01 = load_bT_bf16(Wkb + (size_t)(n0 + 16) * 512 + h * 64, 512);
  v16bf b10 = load_bT_bf16(Wkb + (size_t)n0 * 512 + h * 64 + 32, 512);
  v16bf b11 = load_bT_bf16(Wkb + (size_t)(n0 + 16) * 512 + h * 64 + 32, 512);

  v8f z = {};
  v8f acc0 = z, acc1 = z;
  acc0 = wmma_bf16f32(a0, b00, acc0);
  acc1 = wmma_bf16f32(a0, b01, acc1);
  acc0 = wmma_bf16f32(a1, b10, acc0);
  acc1 = wmma_bf16f32(a1, b11, acc1);

  store_d_bf16(Ub + (size_t)t0 * 2048 + h * 256 + n0, 2048, acc0);
  store_d_bf16(Ub + (size_t)t0 * 2048 + h * 256 + n0 + 16, 2048, acc1);
}

// --------------- K3: fused scores / softmax / w = attn @ C -----------------
// One block per token; wave = head, lane = key index. Context staged to LDS
// as bf16 with row pad 258 (stride 516B -> 129 banks, conflict-free).
__global__ __launch_bounds__(256) void k_attn(const float* __restrict__ ctx,
                                              const unsigned char* __restrict__ mask,
                                              const float* __restrict__ bias,
                                              const __bf16* __restrict__ Ub,
                                              __bf16* __restrict__ Wb) {
  const int t    = blockIdx.x;
  const int tid  = threadIdx.x;
  const int h    = tid >> 5;
  const int lane = tid & 31;

  __shared__ __bf16  Csh[32 * 258];   // 16.5 KB
  __shared__ uint32_t Ush32[1024];    //  4   KB (8 heads x 256 bf16)
  __bf16* Ush = (__bf16*)Ush32;

  const float* cbase = ctx + (size_t)t * 8192;     // [32 x 256] f32
  for (int i = tid; i < 8192; i += 256) {
    int k = i >> 8, c = i & 255;
    Csh[k * 258 + c] = (__bf16)cbase[i];
  }
  const uint32_t* ub = (const uint32_t*)(Ub + (size_t)t * 2048);
  for (int i = tid; i < 1024; i += 256) Ush32[i] = ub[i];
  __syncthreads();

  // s[h, lane] = SCALE * C[lane,:] . u[h,:] + bias
  float s = 0.f;
  const __bf16* crow = Csh + lane * 258;
  const __bf16* urow = Ush + h * 256;
#pragma unroll 8
  for (int c = 0; c < 256; ++c) s += (float)crow[c] * (float)urow[c];
  s = s * 0.125f + bias[t * 32 + lane];            // SCALE = 64^-0.5
  if (!mask[t * 32 + lane]) s = -__builtin_inff();

  // wave32 softmax over k (lanes)
  float m = s;
#pragma unroll
  for (int off = 16; off; off >>= 1) m = fmaxf(m, __shfl_xor(m, off, 32));
  float p = __expf(s - m);
  float sum = p;
#pragma unroll
  for (int off = 16; off; off >>= 1) sum += __shfl_xor(sum, off, 32);
  const float a = p / sum;

  // w[h,c] = sum_k attn[k] * C[k,c]; lanes parallel over c (8 each)
  float wacc[8];
#pragma unroll
  for (int i = 0; i < 8; ++i) wacc[i] = 0.f;
  for (int k = 0; k < 32; ++k) {
    const float ak = __shfl(a, k, 32);
    const __bf16* ck = Csh + k * 258;
#pragma unroll
    for (int cc = 0; cc < 8; ++cc) wacc[cc] += ak * (float)ck[lane + cc * 32];
  }
  __bf16* wrow = Wb + (size_t)t * 2048 + h * 256;
#pragma unroll
  for (int cc = 0; cc < 8; ++cc) wrow[lane + cc * 32] = (__bf16)wacc[cc];
}

// --------------- K4: O_h = W_h @ Wv_h  per head ----------------------------
// grid (256 blocks of 32 tokens, 8 heads); 8 waves = 2(M) x 4(N) tiles.
// A/B ping-pong via unroll-2 across the 8 K-chunks.
__global__ __launch_bounds__(256) void k_gemm_o(const __bf16* __restrict__ Wb,
                                                const __bf16* __restrict__ WvT,
                                                __bf16* __restrict__ Ob) {
  const int w  = threadIdx.x >> 5;
  const int h  = blockIdx.y;
  const int mt = w >> 2, nf = w & 3;
  const int t0 = blockIdx.x * 32 + mt * 16;

  const __bf16* Abase = Wb + (size_t)t0 * 2048 + h * 256;
  const __bf16* Bbase = WvT + (size_t)(h * 64 + nf * 16) * 256;

  v8f acc = {};
  v16bf a0 = load_a_bf16(Abase, 2048);
  v16bf b0 = load_bT_bf16(Bbase, 256);
#pragma unroll
  for (int kk = 0; kk < 256; kk += 64) {
    v16bf a1 = load_a_bf16(Abase + kk + 32, 2048);
    v16bf b1 = load_bT_bf16(Bbase + kk + 32, 256);
    acc = wmma_bf16f32(a0, b0, acc);
    if (kk + 64 < 256) {
      a0 = load_a_bf16(Abase + kk + 64, 2048);
      b0 = load_bT_bf16(Bbase + kk + 64, 256);
    }
    acc = wmma_bf16f32(a1, b1, acc);
  }
  store_d_bf16(Ob + (size_t)t0 * 512 + h * 64 + nf * 16, 512, acc);
}

// --------------- K5: Y = O @ Wo + bo  (f32 out) ----------------------------
__global__ __launch_bounds__(256) void k_gemm_y(const __bf16* __restrict__ Ob,
                                                const __bf16* __restrict__ WoT,
                                                const float* __restrict__ bo,
                                                float* __restrict__ Y) {
  const int w  = threadIdx.x >> 5;
  const int t0 = blockIdx.x * 16;

  __shared__ uint4 Ash4[1024];                     // 16 KB: A tile 16x512 bf16
  __bf16* Ash = (__bf16*)Ash4;
  {
    const uint4* src = (const uint4*)(Ob + (size_t)t0 * 512);
    for (int i = threadIdx.x; i < 1024; i += 256) Ash4[i] = src[i];
  }
  __syncthreads();

  const __bf16* Bbase = WoT + (size_t)(w * 64) * 512;
  v8f z = {};
  v8f acc[4] = {z, z, z, z};

  v16bf b0[4], b1[4];
#pragma unroll
  for (int f = 0; f < 4; ++f) b0[f] = load_bT_bf16(Bbase + (size_t)(f * 16) * 512, 512);

#pragma unroll
  for (int kk = 0; kk < 512; kk += 64) {
#pragma unroll
    for (int f = 0; f < 4; ++f)
      b1[f] = load_bT_bf16(Bbase + (size_t)(f * 16) * 512 + kk + 32, 512);
    v16bf a0 = load_a_bf16(Ash + kk, 512);
#pragma unroll
    for (int f = 0; f < 4; ++f) acc[f] = wmma_bf16f32(a0, b0[f], acc[f]);
    if (kk + 64 < 512) {
#pragma unroll
      for (int f = 0; f < 4; ++f)
        b0[f] = load_bT_bf16(Bbase + (size_t)(f * 16) * 512 + kk + 64, 512);
    }
    v16bf a1 = load_a_bf16(Ash + kk + 32, 512);
#pragma unroll
    for (int f = 0; f < 4; ++f) acc[f] = wmma_bf16f32(a1, b1[f], acc[f]);
  }

  const int lane = threadIdx.x & 31;
  const int n  = lane & 15;
  const int mb = (lane < 16) ? 0 : 8;
#pragma unroll
  for (int f = 0; f < 4; ++f) {
    const int ncol = w * 64 + f * 16;
    const float bv = bo[ncol + n];
    float* D = Y + (size_t)t0 * 512 + ncol;
#pragma unroll
    for (int r = 0; r < 8; ++r) D[(mb + r) * 512 + n] = acc[f][r] + bv;
  }
}

// ---------------------------------------------------------------------------
extern "C" void kernel_launch(void* const* d_in, const int* in_sizes, int n_in,
                              void* d_out, int out_size, void* d_ws, size_t ws_size,
                              hipStream_t stream) {
  (void)in_sizes; (void)n_in; (void)out_size; (void)ws_size;
  const float*         x    = (const float*)d_in[0];          // [8192, 512]
  const float*         ctx  = (const float*)d_in[1];          // [8192, 32, 256]
  const unsigned char* mask = (const unsigned char*)d_in[2];  // [8192, 32] bool
  const float*         bias = (const float*)d_in[3];          // [8192, 32]
  const float*         Wq   = (const float*)d_in[4];          // [512, 512]
  const float*         Wk   = (const float*)d_in[5];          // [256, 512]
  const float*         Wv   = (const float*)d_in[6];          // [256, 512]
  const float*         Wo   = (const float*)d_in[7];          // [512, 512]
  const float*         bo   = (const float*)d_in[8];          // [512]
  float* Y = (float*)d_out;                                   // [8192, 512]

  // workspace layout (bytes), total ~89.5 MB
  char* ws = (char*)d_ws;
  __bf16* Xb  = (__bf16*)(ws + 0);          //  8 MB  [8192 x 512]
  __bf16* Qb  = (__bf16*)(ws + 8388608);    //  8 MB  [8192 x 512]
  __bf16* Ub  = (__bf16*)(ws + 16777216);   // 32 MB  [8192 x 8 x 256]
  __bf16* Wb  = (__bf16*)(ws + 50331648);   // 32 MB  [8192 x 8 x 256]
  __bf16* Ob  = (__bf16*)(ws + 83886080);   //  8 MB  [8192 x 512]
  __bf16* WqT = (__bf16*)(ws + 92274688);   // 512 KB [512 x 512]  (N-major)
  __bf16* Wkb = (__bf16*)(ws + 92798976);   // 256 KB [256 x 512]  (as-is)
  __bf16* WvT = (__bf16*)(ws + 93061120);   // 256 KB [512 x 256]  (inner-major)
  __bf16* WoT = (__bf16*)(ws + 93323264);   // 512 KB [512 x 512]  (N-major)

  // prep: bf16 converts + transposes (weights tiny / L2-resident)
  k_cvt_bf16<<<(4194304 + 255) / 256, 256, 0, stream>>>(x, Xb, 4194304);
  k_transpose_bf16<<<(262144 + 255) / 256, 256, 0, stream>>>(Wq, WqT, 512, 512);
  k_cvt_bf16<<<(131072 + 255) / 256, 256, 0, stream>>>(Wk, Wkb, 131072);
  k_transpose_bf16<<<(131072 + 255) / 256, 256, 0, stream>>>(Wv, WvT, 256, 512);
  k_transpose_bf16<<<(262144 + 255) / 256, 256, 0, stream>>>(Wo, WoT, 512, 512);

  k_gemm_q<<<512, 256, 0, stream>>>(Xb, WqT, Qb);
  k_gemm_u<<<dim3(512, 8), 256, 0, stream>>>(Qb, Wkb, Ub);
  k_attn  <<<8192, 256, 0, stream>>>(ctx, mask, bias, Ub, Wb);
  k_gemm_o<<<dim3(256, 8), 256, 0, stream>>>(Wb, WvT, Ob);
  k_gemm_y<<<512, 256, 0, stream>>>(Ob, WoT, bo, Y);
}
